// BraTSGNN_72670846649168
// MI455X (gfx1250) — matrix operand order repaired and checked by reading, not compiled, not run on hardware
//
#include <hip/hip_runtime.h>
#include <hip/hip_bf16.h>

// ---------------------------------------------------------------------------
// Types for CDNA5 WMMA (wave32): V_WMMA_F32_16X16X32_BF16
// ---------------------------------------------------------------------------
typedef __attribute__((ext_vector_type(16))) __bf16 v16bf;
typedef __attribute__((ext_vector_type(8)))  float  v8f;

__device__ __forceinline__ __bf16 f2bf(float f) {
    union { float f; unsigned u; } v; v.f = f;
    unsigned r = v.u + 0x7FFFu + ((v.u >> 16) & 1u);   // round-to-nearest-even
    union { unsigned short s; __bf16 b; } o;
    o.s = (unsigned short)(r >> 16);
    return o.b;
}

// ---------------------------------------------------------------------------
// Utility kernels
// ---------------------------------------------------------------------------
__global__ void zero_kernel(float* __restrict__ p, size_t n) {
    size_t i = (size_t)blockIdx.x * blockDim.x + threadIdx.x;
    size_t step = (size_t)gridDim.x * blockDim.x;
    for (; i < n; i += step) p[i] = 0.0f;
}

__global__ void degree_kernel(const long long* __restrict__ dst,
                              float* __restrict__ deg, long long E) {
    long long i = (long long)blockIdx.x * blockDim.x + threadIdx.x;
    long long step = (long long)gridDim.x * blockDim.x;
    for (; i < E; i += step)
        atomicAdd(deg + dst[i], 1.0f);
}

__global__ void invdeg_kernel(float* __restrict__ deg, int N) {
    int i = blockIdx.x * blockDim.x + threadIdx.x;
    if (i < N) deg[i] = 1.0f / fmaxf(deg[i], 1.0f);
}

// Gather rows of feat at src, scatter-add into agg at dst. One thread = 4 feats.
template <int D>
__global__ void aggregate_kernel(const float* __restrict__ feat,
                                 const long long* __restrict__ src,
                                 const long long* __restrict__ dst,
                                 float* __restrict__ agg,
                                 long long E) {
    constexpr int CH = D >> 2;                 // D = 12 or 128
    long long total = E * (long long)CH;
    long long i = (long long)blockIdx.x * blockDim.x + threadIdx.x;
    long long step = (long long)gridDim.x * blockDim.x;
    for (; i < total; i += step) {
        long long e = i / CH;
        int c = (int)(i - e * CH);
        long long s = src[e];
        long long d = dst[e];
        const float4 v = *(const float4*)(feat + (size_t)s * D + (size_t)c * 4);
        float* o = agg + (size_t)d * D + (size_t)c * 4;
        __builtin_prefetch(o, 1, 1);           // global_prefetch_b8
        atomicAdd(o + 0, v.x);
        atomicAdd(o + 1, v.y);
        atomicAdd(o + 2, v.z);
        atomicAdd(o + 3, v.w);
    }
}

// ---------------------------------------------------------------------------
// WMMA GEMM: out[N,Nout] = (Aagg*scaleA) @ Wl^T (+ Aroot @ Wr^T) + bias
// A fp32 row-major [N,K]; W row-major [Nout,K]; bf16 compute, fp32 accumulate.
// One wave -> 16 rows x Nout cols.  blockDim = 128 (4 waves).
// Compile-time shape -> straight-line ds_load_b128 + v_wmma stream.
// ---------------------------------------------------------------------------
template <int K>
__device__ __forceinline__ v16bf load_afrag(const float* __restrict__ A,
                                            size_t row, int kb, int kh,
                                            float scale) {
    v16bf a;
    const size_t base = row * (size_t)K;
#pragma unroll
    for (int g = 0; g < 2; ++g) {
        const int bk = kb + kh * 16 + g * 8;   // A layout: lanes 0-15: K {0-15}
        float tmp[8];                          //           lanes16-31: K {16-31}
        if (bk + 8 <= K) {
            const float4 v0 = *(const float4*)(A + base + bk);
            const float4 v1 = *(const float4*)(A + base + bk + 4);
            tmp[0]=v0.x; tmp[1]=v0.y; tmp[2]=v0.z; tmp[3]=v0.w;
            tmp[4]=v1.x; tmp[5]=v1.y; tmp[6]=v1.z; tmp[7]=v1.w;
        } else {
#pragma unroll
            for (int j = 0; j < 8; ++j)
                tmp[j] = (bk + j < K) ? A[base + bk + j] : 0.0f;
        }
#pragma unroll
        for (int j = 0; j < 8; ++j)
            a[g * 8 + j] = f2bf(tmp[j] * scale);
    }
    return a;
}

template <int K, int NTILES, bool FULL, bool RELU>
__global__ __launch_bounds__(128)
void gemm_wmma_kernel(const float* __restrict__ Aagg,
                      const float* __restrict__ scaleA,   // used iff FULL
                      const float* __restrict__ Aroot,    // used iff FULL
                      const float* __restrict__ Wl,
                      const float* __restrict__ Wr,       // used iff FULL
                      const float* __restrict__ bias,
                      float* __restrict__ out,
                      int Nrows) {
    constexpr int KP    = (K + 31) & ~31;                 // 12->32, 128->128
    constexpr int NMAT  = FULL ? 2 : 1;
    constexpr int NOUT  = NTILES * 16;
    __shared__ __align__(32) __bf16 sW[NMAT][NOUT][KP];

    // Stage weights into LDS as zero-padded bf16.
    for (int idx = threadIdx.x; idx < NOUT * KP; idx += blockDim.x) {
        const int n = idx / KP;
        const int k = idx - n * KP;
        sW[0][n][k] = f2bf(k < K ? Wl[(size_t)n * K + k] : 0.0f);
        if (FULL) sW[1][n][k] = f2bf(k < K ? Wr[(size_t)n * K + k] : 0.0f);
    }
    __syncthreads();

    const int wave = threadIdx.x >> 5;
    const int lane = threadIdx.x & 31;
    const int mtile = blockIdx.x * 4 + wave;
    const int Mtiles = Nrows >> 4;                        // Nrows divisible by 16
    if (mtile >= Mtiles) return;                          // wave-uniform

    const int kh  = lane >> 4;
    const int l15 = lane & 31 & 15;
    const size_t row = (size_t)mtile * 16 + l15;
    const float sA = FULL ? scaleA[row] : 1.0f;

    v8f acc[NTILES] = {};

#pragma unroll
    for (int kb = 0; kb < KP; kb += 32) {
        const v16bf aAgg = load_afrag<K>(Aagg, row, kb, kh, sA);
        v16bf aRoot = {};
        if (FULL) aRoot = load_afrag<K>(Aroot, row, kb, kh, 1.0f);
#pragma unroll
        for (int nt = 0; nt < NTILES; ++nt) {
            // B fragment: lane holds col n=nt*16+l15, 16 consecutive K values
            const v16bf b0 = *(const v16bf*)(&sW[0][nt * 16 + l15][kb + kh * 16]);
            acc[nt] = __builtin_amdgcn_wmma_f32_16x16x32_bf16(
                false, aAgg, false, b0, (short)0, acc[nt], false, false);
            if (FULL) {
                const v16bf b1 = *(const v16bf*)(&sW[1][nt * 16 + l15][kb + kh * 16]);
                acc[nt] = __builtin_amdgcn_wmma_f32_16x16x32_bf16(
                    false, aRoot, false, b1, (short)0, acc[nt], false, false);
            }
        }
    }

    // C/D layout: VGPR r -> M = r + 8*kh, N = l15.
#pragma unroll
    for (int nt = 0; nt < NTILES; ++nt) {
        const int col = nt * 16 + l15;
        const float bv = bias[col];
#pragma unroll
        for (int r = 0; r < 8; ++r) {
            const size_t m = (size_t)mtile * 16 + r + kh * 8;
            float v = acc[nt][r] + bv;
            if (RELU) v = fmaxf(v, 0.0f);
            out[m * (size_t)NOUT + col] = v;
        }
    }
}

// ---------------------------------------------------------------------------
// BatchNorm: column sums / sums-of-squares, then fused scale+shift(+relu)
// ---------------------------------------------------------------------------
__global__ void bn_stats_kernel(const float* __restrict__ h,
                                float* __restrict__ stats, int Nrows) {
    const int c = threadIdx.x;                 // blockDim = 128
    const int RPB = 256;
    size_t r0 = (size_t)blockIdx.x * RPB;
    size_t r1 = r0 + RPB; if (r1 > (size_t)Nrows) r1 = (size_t)Nrows;
    float s = 0.0f, s2 = 0.0f;
    for (size_t r = r0; r < r1; ++r) {
        const float v = h[r * 128 + c];
        s += v; s2 += v * v;
    }
    atomicAdd(stats + c, s);
    atomicAdd(stats + 128 + c, s2);
}

__global__ void bn_apply_kernel(float* __restrict__ h,
                                const float* __restrict__ stats,
                                const float* __restrict__ gamma,
                                const float* __restrict__ beta,
                                int Nrows, int relu, float invN) {
    size_t total = (size_t)Nrows * 128;
    size_t i = (size_t)blockIdx.x * blockDim.x + threadIdx.x;
    size_t step = (size_t)gridDim.x * blockDim.x;
    for (; i < total; i += step) {
        const int c = (int)(i & 127);
        const float mean = stats[c] * invN;
        const float var  = stats[128 + c] * invN - mean * mean;
        const float sc = gamma[c] * rsqrtf(var + 1e-5f);
        const float sh = beta[c] - mean * sc;
        float v = h[i] * sc + sh;
        if (relu) v = fmaxf(v, 0.0f);
        h[i] = v;
    }
}

// Final head: logits[n] = bc2 + dot(h1[n, 0:64], Wc2)
__global__ void head2_kernel(const float* __restrict__ h1,
                             const float* __restrict__ w,
                             const float* __restrict__ b,
                             float* __restrict__ out, int Nrows) {
    int n = blockIdx.x * blockDim.x + threadIdx.x;
    int step = gridDim.x * blockDim.x;
    for (; n < Nrows; n += step) {
        float s = b[0];
        const float* hp = h1 + (size_t)n * 64;
#pragma unroll
        for (int j = 0; j < 64; ++j) s += hp[j] * w[j];
        out[n] = s;
    }
}

// ---------------------------------------------------------------------------
// Launch
// ---------------------------------------------------------------------------
extern "C" void kernel_launch(void* const* d_in, const int* in_sizes, int n_in,
                              void* d_out, int out_size, void* d_ws, size_t ws_size,
                              hipStream_t stream) {
    (void)n_in; (void)out_size; (void)ws_size;

    const float*     x   = (const float*)d_in[0];
    const long long* ei  = (const long long*)d_in[1];
    const float*     Wl0 = (const float*)d_in[2];
    const float*     bl0 = (const float*)d_in[3];
    const float*     Wr0 = (const float*)d_in[4];
    const float*     Wl  = (const float*)d_in[5];   // [2,128,128]
    const float*     bl  = (const float*)d_in[6];   // [2,128]
    const float*     Wr  = (const float*)d_in[7];   // [2,128,128]
    const float*     gam = (const float*)d_in[8];   // [3,128]
    const float*     bet = (const float*)d_in[9];   // [3,128]
    const float*     Wc1 = (const float*)d_in[10];  // [64,128]
    const float*     bc1 = (const float*)d_in[11];
    const float*     Wc2 = (const float*)d_in[12];  // [64]
    const float*     bc2 = (const float*)d_in[13];

    const int N = in_sizes[0] / 12;
    const long long E = (long long)in_sizes[1] / 2;
    const long long* src = ei;
    const long long* dst = ei + E;
    const float invN = 1.0f / (float)N;

    float* ws      = (float*)d_ws;
    float* deg     = ws;                                   // [N] -> becomes inv_deg
    float* stats   = ws + (size_t)N;                       // [256]
    float* bufAgg  = stats + 256;                          // [N*128]
    float* bufH0   = bufAgg + (size_t)N * 128;             // [N*128]
    float* bufH1   = bufH0  + (size_t)N * 128;             // [N*128]

    const int T = 256;
    const int gN128  = (int)(((size_t)N * 128 + T - 1) / T);
    const int gE     = (int)((E + T - 1) / T);
    const int Mtiles = N / 16;
    const int gGemm  = (Mtiles + 3) / 4;
    const int gStats = (N + 255) / 256;

    // degree -> inv_deg
    zero_kernel<<<(N + T - 1) / T, T, 0, stream>>>(deg, (size_t)N);
    degree_kernel<<<gE, T, 0, stream>>>(dst, deg, E);
    invdeg_kernel<<<(N + T - 1) / T, T, 0, stream>>>(deg, N);

    // ---- layer 0 (D_IN=12 -> H=128) ----
    zero_kernel<<<(int)(((size_t)N * 12 + T - 1) / T), T, 0, stream>>>(bufAgg, (size_t)N * 12);
    aggregate_kernel<12><<<(int)((E * 3 + T - 1) / T), T, 0, stream>>>(x, src, dst, bufAgg, E);
    gemm_wmma_kernel<12, 8, true, false><<<gGemm, 128, 0, stream>>>(
        bufAgg, deg, x, Wl0, Wr0, bl0, bufH0, N);
    zero_kernel<<<1, T, 0, stream>>>(stats, 256);
    bn_stats_kernel<<<gStats, 128, 0, stream>>>(bufH0, stats, N);
    bn_apply_kernel<<<gN128, T, 0, stream>>>(bufH0, stats, gam + 0, bet + 0, N, 1, invN);

    // ---- layer 1 (H -> H) ----
    zero_kernel<<<gN128, T, 0, stream>>>(bufAgg, (size_t)N * 128);
    aggregate_kernel<128><<<(int)((E * 32 + T - 1) / T), T, 0, stream>>>(bufH0, src, dst, bufAgg, E);
    gemm_wmma_kernel<128, 8, true, false><<<gGemm, 128, 0, stream>>>(
        bufAgg, deg, bufH0, Wl, Wr, bl, bufH1, N);
    zero_kernel<<<1, T, 0, stream>>>(stats, 256);
    bn_stats_kernel<<<gStats, 128, 0, stream>>>(bufH1, stats, N);
    bn_apply_kernel<<<gN128, T, 0, stream>>>(bufH1, stats, gam + 128, bet + 128, N, 1, invN);

    // ---- layer 2 (H -> H, no relu) ----
    zero_kernel<<<gN128, T, 0, stream>>>(bufAgg, (size_t)N * 128);
    aggregate_kernel<128><<<(int)((E * 32 + T - 1) / T), T, 0, stream>>>(bufH1, src, dst, bufAgg, E);
    gemm_wmma_kernel<128, 8, true, false><<<gGemm, 128, 0, stream>>>(
        bufAgg, deg, bufH1, Wl + 128 * 128, Wr + 128 * 128, bl + 128, bufH0, N);
    zero_kernel<<<1, T, 0, stream>>>(stats, 256);
    bn_stats_kernel<<<gStats, 128, 0, stream>>>(bufH0, stats, N);
    bn_apply_kernel<<<gN128, T, 0, stream>>>(bufH0, stats, gam + 256, bet + 256, N, 0, invN);

    // ---- head: [N,128] @ Wc1^T + bc1, relu -> [N,64]; dot Wc2 + bc2 -> [N] ----
    gemm_wmma_kernel<128, 4, false, true><<<gGemm, 128, 0, stream>>>(
        bufH0, nullptr, nullptr, Wc1, nullptr, bc1, bufH1, N);
    head2_kernel<<<(N + T - 1) / T, T, 0, stream>>>(bufH1, Wc2, bc2, (float*)d_out, N);
}
